// WatNet2D_9440338117124
// MI455X (gfx1250) — matrix operand order repaired and checked by reading, not compiled, not run on hardware
//
#include <hip/hip_runtime.h>
#include <hip/hip_bf16.h>

// CDNA5 / gfx1250 implementation of WatNet2D forward.
// Every convolution runs as implicit GEMM on v_wmma_f32_16x16x32_f16
// (f16 inputs staged through LDS, f32 accumulation).

typedef __attribute__((ext_vector_type(16))) _Float16 v16h;
typedef __attribute__((ext_vector_type(8)))  float    v8f;

#define TB 256   // 8 wave32 waves
#define KT 32    // K per WMMA step (f16 16x16x32)
#define MT 32    // Cout tile per block (2 waves in M)
#define NT 128   // pixel tile per block (4 waves in N, 2 sub-tiles each)

// pack two f32 -> two f16 in one dword (for ds_store_b32 staging)
static __device__ __forceinline__ unsigned pack2(float a, float b)
{
    union { _Float16 h[2]; unsigned u; } t;
    t.h[0] = (_Float16)a;
    t.h[1] = (_Float16)b;
    return t.u;
}

// assemble a v16h fragment from two 16B LDS chunks (ds_load_b128 x2)
static __device__ __forceinline__ v16h bc16(uint4 a, uint4 b)
{
    union { unsigned u[8]; v16h h; } t;
    t.u[0] = a.x; t.u[1] = a.y; t.u[2] = a.z; t.u[3] = a.w;
    t.u[4] = b.x; t.u[5] = b.y; t.u[6] = b.z; t.u[7] = b.w;
    return t.h;
}

// Branchless im2col fetch. The load is from a clamped (always valid) address
// and the result is multiplied by a 0/1 mask: the value is live on both
// paths so the compiler cannot sink the load into a divergent branch.
template<int KH, int KW>
static __device__ __forceinline__ float fetch_x(const float* __restrict__ xb,
    int kg, int Ktot, int Cin, int HWp, int stride_oh_pad, int stride_ow_pad,
    int H, int W, float nmask)
{
    constexpr int KK = KH * KW;
    int ci = kg / KK;                 // compile-time-constant divisors
    int kr = kg - ci * KK;
    int kh = kr / KW;
    int kw = kr - kh * KW;
    int ih = stride_oh_pad + kh;      // oh*stride - pad + kh
    int iw = stride_ow_pad + kw;
    float mask = (kg < Ktot && ih >= 0 && ih < H && iw >= 0 && iw < W) ? nmask : 0.0f;
    int cic = ci < Cin ? ci : Cin - 1;
    int ihc = ih < 0 ? 0 : (ih >= H ? H - 1 : ih);
    int iwc = iw < 0 ? 0 : (iw >= W ? W - 1 : iw);
    float v = xb[cic * HWp + ihc * W + iwc];   // unconditional load
    return v * mask;
}

// ---------------------------------------------------------------------------
// Implicit-GEMM conv: M = Cout, N = B*Ho*Wo, K = Cin*KH*KW.
// 256 threads = 8 waves laid out 2(M) x 4(N); each wave: one A fragment,
// two B fragments, two WMMAs per K step (16x32 output per wave).
// ---------------------------------------------------------------------------
template<int KH, int KW>
__global__ __launch_bounds__(TB)
void conv2d_wmma(const float* __restrict__ x, const float* __restrict__ wgt,
                 const float* __restrict__ bias, float* __restrict__ y,
                 int B, int Cin, int H, int W, int Cout, int Ho, int Wo,
                 int stride, int pad, int cin_off, int cin_tot, int do_relu)
{
    constexpr int KK = KH * KW;
    __shared__ alignas(16) _Float16 As[MT][KT + 8];   // M x K (K contiguous)
    __shared__ alignas(16) _Float16 Bs[NT][KT + 8];   // N x K (K contiguous)

    const int tid  = threadIdx.x;
    const int lane = tid & 31;            // wave32 lane
    const int wave = tid >> 5;
    const int wm   = wave >> 2;           // 0..1  (M sub-tile)
    const int wn   = wave & 3;            // 0..3  (N quad, 32 cols each)
    const int g    = lane >> 4;           // half-wave group
    const int lr   = lane & 15;

    const int HW   = Ho * Wo;
    const int HWp  = H * W;
    const long long Npix  = (long long)B * HW;
    const long long tileN = (long long)blockIdx.x * NT;
    const int tileM = blockIdx.y * MT;
    const int Ktot  = Cin * KK;

    // ---- K-invariant per-thread staging state ----
    // A: 2 dwords/thread of the 32x16-dword tile
    const float* baseA[2];
    float amask[2];
    int akd[2], am[2];
    #pragma unroll
    for (int j = 0; j < 2; ++j) {
        int i  = tid + j * TB;
        int m  = i >> 4;
        int kd = i & 15;
        am[j] = m; akd[j] = kd;
        int mg  = tileM + m;
        int mgc = mg < Cout ? mg : Cout - 1;
        baseA[j] = wgt + (long long)mgc * Ktot;
        amask[j] = (mg < Cout) ? 1.0f : 0.0f;
    }
    // B: 8 dwords/thread of the 128x16-dword tile; pixel decode is fixed
    const float* baseB[8];
    float nmask[8];
    int nj[8], kdj[8], sohj[8], sowj[8];
    #pragma unroll
    for (int j = 0; j < 8; ++j) {
        int i  = tid + j * TB;
        int n  = i >> 4;
        int kd = i & 15;
        nj[j] = n; kdj[j] = kd;
        long long ng = tileN + n;
        bool nv = ng < Npix;
        long long ngc = nv ? ng : 0;
        int b = (int)(ngc / HW);
        int r = (int)(ngc - (long long)b * HW);
        int oh = r / Wo;
        int ow = r - oh * Wo;
        baseB[j] = x + ((long long)b * cin_tot + cin_off) * HWp;
        sohj[j] = oh * stride - pad;
        sowj[j] = ow * stride - pad;
        nmask[j] = nv ? 1.0f : 0.0f;
    }

    v8f acc0 = {}, acc1 = {};

    for (int kt = 0; kt < Ktot; kt += KT) {
        // ---- stage A (weights): branchless masked loads ----
        #pragma unroll
        for (int j = 0; j < 2; ++j) {
            int kg  = kt + 2 * akd[j];
            int k0c = kg     < Ktot ? kg     : Ktot - 1;
            int k1c = kg + 1 < Ktot ? kg + 1 : Ktot - 1;
            float v0 = baseA[j][k0c] * ((kg     < Ktot) ? amask[j] : 0.0f);
            float v1 = baseA[j][k1c] * ((kg + 1 < Ktot) ? amask[j] : 0.0f);
            *(unsigned*)&As[am[j]][2 * akd[j]] = pack2(v0, v1);
        }
        // ---- stage B (im2col): branchless masked loads ----
        #pragma unroll
        for (int j = 0; j < 8; ++j) {
            int kg = kt + 2 * kdj[j];
            float v0 = fetch_x<KH, KW>(baseB[j], kg,     Ktot, Cin, HWp,
                                       sohj[j], sowj[j], H, W, nmask[j]);
            float v1 = fetch_x<KH, KW>(baseB[j], kg + 1, Ktot, Cin, HWp,
                                       sohj[j], sowj[j], H, W, nmask[j]);
            *(unsigned*)&Bs[nj[j]][2 * kdj[j]] = pack2(v0, v1);
        }
        // L2 prefetch of next weight K-slab (global_prefetch_b8)
        if (kt + KT < Ktot && wave == 0 && (tileM + lane) < Cout)
            __builtin_prefetch(&wgt[(long long)(tileM + lane) * Ktot + kt + KT], 0, 3);

        __syncthreads();

        // ---- fragments: vectorized LDS reads (ds_load_b128) ----
        // A 16x32: lane holds row M=lr, halves g*16 .. g*16+15 (contiguous)
        const uint4* pa = (const uint4*)&As[wm * 16 + lr][0];
        v16h afrag = bc16(pa[2 * g], pa[2 * g + 1]);

        // B 32x16: lane holds col N, halves g*16 .. g*16+15 (contiguous)
        const uint4* pb0 = (const uint4*)&Bs[wn * 32 + lr][0];
        v16h bfrag0 = bc16(pb0[2 * g], pb0[2 * g + 1]);
        const uint4* pb1 = (const uint4*)&Bs[wn * 32 + 16 + lr][0];
        v16h bfrag1 = bc16(pb1[2 * g], pb1[2 * g + 1]);

        // EXEC all-ones here (no divergent branch encloses the WMMAs)
        acc0 = __builtin_amdgcn_wmma_f32_16x16x32_f16(
                   false, afrag, false, bfrag0, (short)0, acc0, false, false);
        acc1 = __builtin_amdgcn_wmma_f32_16x16x32_f16(
                   false, afrag, false, bfrag1, (short)0, acc1, false, false);

        __syncthreads();
    }

    // ---- epilogue: C layout = VGPR v -> M = v + 8*g, N = lane&15 ----
    #pragma unroll
    for (int t = 0; t < 2; ++t) {
        long long ng = tileN + wn * 32 + t * 16 + lr;
        if (ng < Npix) {
            int b = (int)(ng / HW);
            int r = (int)(ng - (long long)b * HW);
            #pragma unroll
            for (int vv = 0; vv < 8; ++vv) {
                int mg = tileM + wm * 16 + g * 8 + vv;
                if (mg < Cout) {
                    float val = (t == 0 ? acc0[vv] : acc1[vv]) + bias[mg];
                    if (do_relu) val = fmaxf(val, 0.0f);
                    y[((long long)b * Cout + mg) * HW + r] = val;
                }
            }
        }
    }
}

// ---------------------------------------------------------------------------
// Haar decomposition: emits cA (optional) and packed detail channels.
// mode4=0: pack [cH,cV,cD] as channel c*3+j ; mode4=1: pack [cA,cH,cV,cD].
// ---------------------------------------------------------------------------
__global__ void haar_kernel(const float* __restrict__ x, float* __restrict__ cA,
                            float* __restrict__ pack,
                            int B, int C, int H, int W, int mode4)
{
    int Ho = H >> 1, Wo = W >> 1;
    long long total = (long long)B * C * Ho * Wo;
    long long idx = (long long)blockIdx.x * TB + threadIdx.x;
    if (idx >= total) return;
    int wo = (int)(idx % Wo);
    long long r = idx / Wo;
    int ho = (int)(r % Ho); r /= Ho;
    int c  = (int)(r % C);
    int b  = (int)(r / C);

    const float* base = x + (((long long)b * C + c) * H + 2 * ho) * W + 2 * wo;
    float p = base[0], q = base[1], rr = base[W], s = base[W + 1];
    float A  = (p + q + rr + s) * 0.5f;
    float Hc = (p + q - rr - s) * 0.5f;
    float V  = (p - q + rr - s) * 0.5f;
    float D  = (p - q - rr + s) * 0.5f;

    long long plane = (long long)Ho * Wo;
    long long pix   = (long long)ho * Wo + wo;
    if (cA) cA[((long long)b * C + c) * plane + pix] = A;
    int nc = mode4 ? 4 : 3;
    float* wb = pack + ((long long)b * C * nc + (long long)c * nc) * plane + pix;
    if (mode4) {
        wb[0] = A; wb[plane] = Hc; wb[2 * plane] = V; wb[3 * plane] = D;
    } else {
        wb[0] = Hc; wb[plane] = V; wb[2 * plane] = D;
    }
}

// wat gate: out = relu(x*w + w)
__global__ void watmul_kernel(const float* __restrict__ x, const float* __restrict__ w,
                              float* __restrict__ y, long long n)
{
    long long i = (long long)blockIdx.x * TB + threadIdx.x;
    if (i >= n) return;
    float wv = w[i];
    y[i] = fmaxf(fmaf(x[i], wv, wv), 0.0f);
}

// g(80ch) += concat([x(64ch), h[:,0:16]])
__global__ void addcat_kernel(float* __restrict__ gbuf, const float* __restrict__ x,
                              const float* __restrict__ h, int B, int HW)
{
    long long total = (long long)B * 80 * HW;
    long long idx = (long long)blockIdx.x * TB + threadIdx.x;
    if (idx >= total) return;
    int p = (int)(idx % HW);
    long long r = idx / HW;
    int c = (int)(r % 80);
    int b = (int)(r / 80);
    float add = (c < 64) ? x[((long long)b * 64 + c) * HW + p]
                         : h[((long long)b * 64 + (c - 64)) * HW + p];
    gbuf[idx] += add;
}

// Transposed conv, stride 2, 4x4, 64->64.  y[n,o,oh,ow] = sum over parity-valid
// taps of x[n,i,ih,iw] * w[i,o,3-kh,3-kw]  (w given as (in=64,out=64,4,4)).
__global__ void convT_kernel(const float* __restrict__ x, const float* __restrict__ w,
                             const float* __restrict__ bias, float* __restrict__ y,
                             int B, int C, int Hin, int Win)
{
    int Ho = 2 * Hin, Wo = 2 * Win;
    long long total = (long long)B * C * Ho * Wo;
    long long idx = (long long)blockIdx.x * TB + threadIdx.x;
    if (idx >= total) return;
    int ow = (int)(idx % Wo);
    long long r = idx / Wo;
    int oh = (int)(r % Ho); r /= Ho;
    int co = (int)(r % C);
    int b  = (int)(r / C);

    float acc = bias[co];
    for (int kh = 0; kh < 4; ++kh) {
        int j = oh - 2 + kh;
        if (j < 0 || j > 2 * Hin - 2 || (j & 1)) continue;
        int ih = j >> 1;
        for (int kw = 0; kw < 4; ++kw) {
            int jj = ow - 2 + kw;
            if (jj < 0 || jj > 2 * Win - 2 || (jj & 1)) continue;
            int iw = jj >> 1;
            const float* xp = x + (((long long)b * C) * Hin + ih) * Win + iw;
            const float* wp = w + (long long)co * 16 + (3 - kh) * 4 + (3 - kw);
            long long xstep = (long long)Hin * Win;
            for (int ci = 0; ci < 64; ++ci)
                acc = fmaf(xp[ci * xstep], wp[(long long)ci * 64 * 16], acc);
        }
    }
    y[idx] = acc;
}

// out[b,c,:] = t[b,0,:] + x[b,c,:]  (broadcast over 3 channels)
__global__ void finaladd_kernel(const float* __restrict__ t, const float* __restrict__ x,
                                float* __restrict__ out, int B, int HW)
{
    long long total = (long long)B * 3 * HW;
    long long idx = (long long)blockIdx.x * TB + threadIdx.x;
    if (idx >= total) return;
    int p = (int)(idx % HW);
    long long r = idx / HW;
    int b = (int)(r / 3);
    out[idx] = t[(long long)b * HW + p] + x[idx];
}

// ---------------------------------------------------------------------------
// Host-side orchestration
// ---------------------------------------------------------------------------
struct WB { const float* w; const float* b; };

static inline void launch_conv(hipStream_t s, const float* x, const float* w,
                               const float* b, float* y,
                               int B, int Cin, int H, int W, int Cout,
                               int Kh, int Kw, int stride, int pad,
                               int cin_off, int cin_tot, int relu)
{
    int Ho = (H + 2 * pad - Kh) / stride + 1;
    int Wo = (W + 2 * pad - Kw) / stride + 1;
    long long Npix = (long long)B * Ho * Wo;
    dim3 grid((unsigned)((Npix + NT - 1) / NT), (unsigned)((Cout + MT - 1) / MT));
    if (Kh == 3)
        conv2d_wmma<3, 3><<<grid, TB, 0, s>>>(x, w, b, y, B, Cin, H, W, Cout, Ho, Wo,
                                              stride, pad, cin_off, cin_tot, relu);
    else if (Kh == 1)
        conv2d_wmma<1, 1><<<grid, TB, 0, s>>>(x, w, b, y, B, Cin, H, W, Cout, Ho, Wo,
                                              stride, pad, cin_off, cin_tot, relu);
    else
        conv2d_wmma<17, 17><<<grid, TB, 0, s>>>(x, w, b, y, B, Cin, H, W, Cout, Ho, Wo,
                                                stride, pad, cin_off, cin_tot, relu);
}

static void run_dblock(hipStream_t s, const float* x, float* out, const WB* p,
                       float* P, float* Q, float* Hb,
                       int B, int H, int W, int stride, int final_relu)
{
    // bs chain
    launch_conv(s, x,  p[0].w, p[0].b, P,  B, 64, H, W, 48, 3, 3, 1, 1, 0, 64, 1);
    launch_conv(s, P,  p[1].w, p[1].b, Q,  B, 48, H, W, 32, 3, 3, 1, 1, 0, 48, 1);
    launch_conv(s, Q,  p[2].w, p[2].b, Hb, B, 32, H, W, 64, 3, 3, 1, 1, 0, 32, 1);
    // as chain on h[:,16:64]
    launch_conv(s, Hb, p[3].w, p[3].b, P,  B, 48, H, W, 64, 3, 3, 1, 1, 16, 64, 1);
    launch_conv(s, P,  p[4].w, p[4].b, Q,  B, 64, H, W, 48, 3, 3, 1, 1, 0, 64, 1);
    launch_conv(s, Q,  p[5].w, p[5].b, P,  B, 48, H, W, 80, 3, 3, 1, 1, 0, 48, 1);
    // g += concat([x, h16])
    long long n = (long long)B * 80 * H * W;
    addcat_kernel<<<(unsigned)((n + TB - 1) / TB), TB, 0, s>>>(P, x, Hb, B, H * W);
    // 1x1 down conv (stride 1 or 2), optional relu
    launch_conv(s, P, p[6].w, p[6].b, out, B, 80, H, W, 64, 1, 1, stride, 0, 0, 80, final_relu);
}

static void run_wat(hipStream_t s, const float* x, const float* wp, float* out,
                    const WB* c, float* P, float* Q, int B, int Cw, int H, int W)
{
    launch_conv(s, wp, c[0].w, c[0].b, P, B, Cw, H, W, 32, 3, 3, 1, 1, 0, Cw, 1);
    launch_conv(s, P,  c[1].w, c[1].b, Q, B, 32, H, W, 64, 3, 3, 1, 1, 0, 32, 0);
    long long n = (long long)B * 64 * H * W;
    watmul_kernel<<<(unsigned)((n + TB - 1) / TB), TB, 0, s>>>(x, Q, out, n);
}

extern "C" void kernel_launch(void* const* d_in, const int* in_sizes, int n_in,
                              void* d_out, int out_size, void* d_ws, size_t ws_size,
                              hipStream_t stream)
{
    (void)in_sizes; (void)n_in; (void)out_size; (void)ws_size;
    const int B = 4;

    // Flattened input order (setup_inputs insertion order, tuples as (w,b)):
    // 0:x 1-2:c3_1 3-4:c3_2 5-18:d1{bs1,bs2,bs3,as1,as2,as3,down}
    // 19-22:wat12{c1,c2} 23-26:wat3{c1,c2} 27-40:inter 41-54:recon
    // 55-56:recon_up 57-58:final
    auto F = [&](int i) { return (const float*)d_in[i]; };
    const float* x = F(0);
    WB d1p[7], interp[7], reconp[7];
    for (int j = 0; j < 7; ++j) {
        d1p[j]    = { F(5 + 2 * j),  F(6 + 2 * j) };
        interp[j] = { F(27 + 2 * j), F(28 + 2 * j) };
        reconp[j] = { F(41 + 2 * j), F(42 + 2 * j) };
    }
    WB wat12p[2] = { { F(19), F(20) }, { F(21), F(22) } };
    WB wat3p[2]  = { { F(23), F(24) }, { F(25), F(26) } };
    const float* upw = F(55); const float* upb = F(56);
    const float* fiw = F(57); const float* fib = F(58);

    // Workspace layout: 5 rotating buffers of 80ch @ 256x256 + wavelet bufs.
    float* ws = (float*)d_ws;
    const long long S = (long long)B * 256 * 256;   // one channel plane @256
    float* bufX = ws;
    float* bufO = bufX + 80 * S;
    float* bufP = bufO + 80 * S;
    float* bufQ = bufP + 80 * S;
    float* bufH = bufQ + 80 * S;
    float* cA1  = bufH + 80 * S;          // 3ch @128
    float* w1   = cA1 + 3 * (S / 4);      // 9ch @128
    float* cA2  = w1 + 9 * (S / 4);       // 3ch @64
    float* w2   = cA2 + 3 * (S / 16);     // 9ch @64
    float* w3   = w2 + 9 * (S / 16);      // 12ch @32

    hipStream_t s = stream;
    auto blocks = [](long long n) { return (unsigned)((n + TB - 1) / TB); };

    // Wavelet pyramid from x (3 channels)
    haar_kernel<<<blocks((long long)B * 3 * 128 * 128), TB, 0, s>>>(x,   cA1, w1, B, 3, 256, 256, 0);
    haar_kernel<<<blocks((long long)B * 3 * 64 * 64),   TB, 0, s>>>(cA1, cA2, w2, B, 3, 128, 128, 0);
    haar_kernel<<<blocks((long long)B * 3 * 32 * 32),   TB, 0, s>>>(cA2, (float*)nullptr, w3, B, 3, 64, 64, 1);

    // Stem
    launch_conv(s, x,    F(1), F(2), bufP, B, 3,  256, 256, 64, 3, 3, 1, 1, 0, 3,  1);
    launch_conv(s, bufP, F(3), F(4), bufX, B, 64, 256, 256, 64, 3, 3, 1, 1, 0, 64, 1);

    // Downsampling path with wavelet gates
    run_dblock(s, bufX, bufO, d1p, bufP, bufQ, bufH, B, 256, 256, 2, 0);   // ->128
    run_wat(s, bufO, w1, bufX, wat12p, bufP, bufQ, B, 9, 128, 128);
    run_dblock(s, bufX, bufO, d1p, bufP, bufQ, bufH, B, 128, 128, 2, 0);   // ->64
    run_wat(s, bufO, w2, bufX, wat12p, bufP, bufQ, B, 9, 64, 64);
    run_dblock(s, bufX, bufO, d1p, bufP, bufQ, bufH, B, 64, 64, 2, 0);     // ->32
    run_wat(s, bufO, w3, bufX, wat3p, bufP, bufQ, B, 12, 32, 32);

    // Intermediate blocks @32
    run_dblock(s, bufX, bufO, interp, bufP, bufQ, bufH, B, 32, 32, 1, 1);
    run_dblock(s, bufO, bufX, interp, bufP, bufQ, bufH, B, 32, 32, 1, 1);

    // Reconstruction: 3x (dblock + convT upsample)  32->64->128->256
    int hw = 32;
    for (int it = 0; it < 3; ++it) {
        run_dblock(s, bufX, bufO, reconp, bufP, bufQ, bufH, B, hw, hw, 1, 1);
        long long n = (long long)B * 64 * (2 * hw) * (2 * hw);
        convT_kernel<<<blocks(n), TB, 0, s>>>(bufO, upw, upb, bufX, B, 64, hw, hw);
        hw *= 2;
    }

    // Final 17x17 conv (64->1, pad 8) and broadcast residual add
    launch_conv(s, bufX, fiw, fib, bufP, B, 64, 256, 256, 1, 17, 17, 1, 8, 0, 64, 0);
    long long n2 = (long long)B * 3 * 256 * 256;
    finaladd_kernel<<<blocks(n2), TB, 0, s>>>(bufP, x, (float*)d_out, B, 256 * 256);
}
